// GPT_10307921511058
// MI455X (gfx1250) — compile-verified, matrix operand order checked by reading, not run on hardware
//
#include <hip/hip_runtime.h>
#include <hip/hip_bf16.h>

#define BB 4
#define DD 512
#define HH 8
#define HDIM 64
#define VV 1024
#define SS 1792
#define NBK 28
#define LL 2
#define DFF 2048
#define MT (BB * SS)   /* 7168 */

typedef __attribute__((ext_vector_type(16))) __bf16 bf16x16;
typedef __attribute__((ext_vector_type(8)))  __bf16 bf16x8;
typedef __attribute__((ext_vector_type(8)))  float  f32x8;
typedef __attribute__((ext_vector_type(4)))  float  f32x4;

// ---------------------------------------------------------------------------
// WMMA helpers
// ---------------------------------------------------------------------------
__device__ __forceinline__ f32x8 wmma_bf16(bf16x16 a, bf16x16 b, f32x8 c) {
  return __builtin_amdgcn_wmma_f32_16x16x32_bf16(
      /*neg_a=*/false, a, /*neg_b=*/false, b,
      /*c_mod=*/(short)0, c, /*reuse_a=*/false, /*reuse_b=*/false);
}

// Load a 16x32 bf16 WMMA fragment (A layout; B identical when the tile is
// stored [N][K]) from an LDS tile stored row-major with `stride` elements/row.
//   lanes 0-15 : row = lane,    K = k0+{0..7}   and k0+{16..23}
//   lanes 16-31: row = lane-16, K = k0+{8..15}  and k0+{24..31}
__device__ __forceinline__ bf16x16 load_frag(const __bf16* base, int row_base,
                                             int stride, int k0, int lane) {
  const __bf16* p = base + (row_base + (lane & 15)) * stride + k0 + ((lane >> 4) << 3);
  bf16x8 lo = *(const bf16x8*)p;
  bf16x8 hi = *(const bf16x8*)(p + 16);
  return __builtin_shufflevector(lo, hi, 0, 1, 2, 3, 4, 5, 6, 7,
                                 8, 9, 10, 11, 12, 13, 14, 15);
}

// gfx1250 async copy: 16B global -> 16B LDS, tracked by ASYNCcnt (ISA §10.7).
// VDST = per-lane LDS byte offset (low 32 bits of the flat shared pointer,
// per the LDS aperture truncation rule), VADDR = 64-bit global address.
__device__ __forceinline__ void async_g2lds_b128(void* lds_ptr, const void* gptr) {
  unsigned int lds_off = (unsigned int)(unsigned long long)lds_ptr;
  asm volatile("global_load_async_to_lds_b128 %0, %1, off"
               :: "v"(lds_off), "v"(gptr) : "memory");
}
__device__ __forceinline__ void async_wait0() {
  asm volatile("s_wait_asynccnt 0x0" ::: "memory");
}

// ---------------------------------------------------------------------------
// Weight cast: fp32 [K][N] row-major  ->  bf16 [N][K] row-major (transposed)
// ---------------------------------------------------------------------------
__global__ void cast_transpose_kernel(const float* __restrict__ W,
                                      __bf16* __restrict__ out, int K, int N) {
  int idx = blockIdx.x * blockDim.x + threadIdx.x;
  if (idx >= K * N) return;
  int n = idx % N;
  int k = idx / N;
  out[(size_t)n * K + k] = (__bf16)W[idx];
}

// ---------------------------------------------------------------------------
// Embedding gather: h[b,s,:] = tok_emb[token] + pos_emb[s]
// ---------------------------------------------------------------------------
__global__ void embed_kernel(const int* __restrict__ x_tok,
                             const int* __restrict__ c_tok,
                             const float* __restrict__ x_emb,
                             const float* __restrict__ c_emb,
                             const float* __restrict__ x_pos,
                             const float* __restrict__ c_pos,
                             float* __restrict__ h) {
  int row = blockIdx.x;          // 0 .. B*S-1
  int b = row / SS, s = row % SS;
  const float* e;
  const float* p;
  if (s < 256) {
    e = c_emb + (size_t)c_tok[b * 256 + s] * DD;
    p = c_pos + (size_t)s * DD;
  } else {
    int si = s - 256;
    e = x_emb + (size_t)x_tok[b * 1536 + si] * DD;
    p = x_pos + (size_t)si * DD;
  }
  float* o = h + (size_t)row * DD;
  for (int d = threadIdx.x; d < DD; d += blockDim.x) o[d] = e[d] + p[d];
}

// ---------------------------------------------------------------------------
// LayerNorm: one wave32 per row of D=512 (16 elements / lane)
// ---------------------------------------------------------------------------
__global__ __launch_bounds__(128) void ln_kernel(const float* __restrict__ x,
                                                 const float* __restrict__ g,
                                                 const float* __restrict__ b,
                                                 float* __restrict__ y) {
  int wave = threadIdx.x >> 5, lane = threadIdx.x & 31;
  int row = blockIdx.x * 4 + wave;
  const float* xr = x + (size_t)row * DD;
  float vals[16];
  float sum = 0.f, sum2 = 0.f;
#pragma unroll
  for (int i = 0; i < 16; i++) {
    float v = xr[lane + i * 32];
    vals[i] = v;
    sum += v;
    sum2 += v * v;
  }
#pragma unroll
  for (int m = 16; m >= 1; m >>= 1) {
    sum += __shfl_xor(sum, m, 32);
    sum2 += __shfl_xor(sum2, m, 32);
  }
  float mean = sum * (1.0f / DD);
  float var = sum2 * (1.0f / DD) - mean * mean;
  float inv = rsqrtf(var + 1e-5f);
  float* yr = y + (size_t)row * DD;
#pragma unroll
  for (int i = 0; i < 16; i++) {
    int d = lane + i * 32;
    yr[d] = (vals[i] - mean) * inv * g[d] + b[d];
  }
}

// ---------------------------------------------------------------------------
// bf16 WMMA GEMM: C[M,N] = act(A[M,K] @ Bt^T + bias) (+ C residual)
//   A: fp32 row-major (cast to bf16 during the LDS stage)
//   Bt: bf16, pre-transposed [N][K]
// Block tile 128x128, 8 waves as 4(M)x2(N); wave tile 32x64 = 8 accumulators,
// 8 WMMAs per 32-K step.  2-stage software pipeline: global->reg staging of
// slice k+1 overlaps WMMA on slice k.  ACT: 0 none, 1 tanh-GELU.
// RES: C += result (f32 out only).  HB: bias.  OBF: store bf16 output.
// ---------------------------------------------------------------------------
template <int ACT, int RES, bool HB, bool OBF>
__global__ __launch_bounds__(256) void gemm_kernel(
    const float* __restrict__ A, const __bf16* __restrict__ Bt,
    const float* __restrict__ bias, void* __restrict__ Cv, int M, int N,
    int K) {
  __shared__ __align__(16) __bf16 sA[128][32];
  __shared__ __align__(16) __bf16 sB[128][32];
  int wave = threadIdx.x >> 5, lane = threadIdx.x & 31;
  int wm = wave >> 1;  // 0..3 : 32-row M subtile
  int wn = wave & 1;   // 0..1 : 64-col N subtile
  int m0 = blockIdx.y * 128;
  int n0 = blockIdx.x * 128;

  f32x8 acc[2][4];
#pragma unroll
  for (int mt = 0; mt < 2; mt++)
#pragma unroll
    for (int nt = 0; nt < 4; nt++) acc[mt][nt] = (f32x8){0, 0, 0, 0, 0, 0, 0, 0};

  // staging: each thread owns 16 A floats and 16 B bf16 of the 32-wide slice
  int sr = threadIdx.x >> 1;          // row 0..127 (A rows / B "n" rows)
  int scol = (threadIdx.x & 1) * 16;  // col 0 or 16
  const float* gA = A + (size_t)(m0 + sr) * K + scol;
  const __bf16* gB = Bt + (size_t)(n0 + sr) * K + scol;

  f32x4 ra0, ra1, ra2, ra3;
  bf16x8 rb0, rb1;
  auto stage_load = [&](int k0) {
    const float* pa = gA + k0;
    ra0 = *(const f32x4*)(pa + 0);
    ra1 = *(const f32x4*)(pa + 4);
    ra2 = *(const f32x4*)(pa + 8);
    ra3 = *(const f32x4*)(pa + 12);
    const __bf16* pb = gB + k0;
    rb0 = *(const bf16x8*)(pb + 0);
    rb1 = *(const bf16x8*)(pb + 8);
  };
  auto stage_store = [&]() {
    __bf16* da = &sA[sr][scol];
#pragma unroll
    for (int j = 0; j < 4; j++) {
      da[j] = (__bf16)ra0[j];
      da[4 + j] = (__bf16)ra1[j];
      da[8 + j] = (__bf16)ra2[j];
      da[12 + j] = (__bf16)ra3[j];
    }
    *(bf16x8*)&sB[sr][scol] = rb0;
    *(bf16x8*)&sB[sr][scol + 8] = rb1;
  };

  stage_load(0);
  for (int k0 = 0; k0 < K; k0 += 32) {
    __syncthreads();  // all waves finished reading LDS slice k0-32
    stage_store();
    __syncthreads();
    if (k0 + 32 < K) {
      stage_load(k0 + 32);
      if (k0 + 64 < K) {  // gfx1250 global_prefetch two slices ahead
        __builtin_prefetch(gA + k0 + 64, 0, 0);
        __builtin_prefetch(gB + k0 + 64, 0, 0);
      }
    }
    bf16x16 af0 = load_frag(&sA[0][0], wm * 32, 32, 0, lane);
    bf16x16 af1 = load_frag(&sA[0][0], wm * 32 + 16, 32, 0, lane);
#pragma unroll
    for (int nt = 0; nt < 4; nt++) {
      bf16x16 bf = load_frag(&sB[0][0], wn * 64 + nt * 16, 32, 0, lane);
      acc[0][nt] = wmma_bf16(af0, bf, acc[0][nt]);
      acc[1][nt] = wmma_bf16(af1, bf, acc[1][nt]);
    }
  }

  // Epilogue (branch-free per template): C layout lane/component mapping
  int hi8 = (lane >> 4) * 8;
  int cl = lane & 15;
#pragma unroll
  for (int mt = 0; mt < 2; mt++) {
#pragma unroll
    for (int nt = 0; nt < 4; nt++) {
      int n = n0 + wn * 64 + nt * 16 + cl;
      float bn = HB ? bias[n] : 0.f;
#pragma unroll
      for (int r = 0; r < 8; r++) {
        int m = m0 + wm * 32 + mt * 16 + r + hi8;
        float vl = acc[mt][nt][r] + bn;
        if (ACT == 1) {  // tanh-approx GELU (jax.nn.gelu default)
          float x = vl;
          float t = tanhf(0.7978845608028654f * (x + 0.044715f * x * x * x));
          vl = 0.5f * x * (1.0f + t);
        }
        size_t idx = (size_t)m * N + n;
        if (OBF) {
          ((__bf16*)Cv)[idx] = (__bf16)vl;
        } else {
          float* C = (float*)Cv;
          if (RES) vl += C[idx];
          C[idx] = vl;
        }
      }
    }
  }
}

// ---------------------------------------------------------------------------
// Block-sparse flash attention.  q/k/v are bf16 [B,S,D] (= [B,S,H,HD] view).
// Grid = (NB, H, B); 4 waves; wave w owns query rows [qi*64+w*16, +16).
// scores = (q k^T * scale + bias) * mask over layout-active 64x64 blocks,
// online softmax, ctx accumulated with WMMA, h += ctx (reference residual).
// K tiles staged with gfx1250 GLOBAL_LOAD_ASYNC_TO_LDS_B128 (ASYNCcnt).
// ---------------------------------------------------------------------------
__global__ __launch_bounds__(128) void attn_kernel(
    const __bf16* __restrict__ q, const __bf16* __restrict__ k,
    const __bf16* __restrict__ v, const int* __restrict__ layout,
    const float* __restrict__ bias, const float* __restrict__ mask,
    float* __restrict__ h) {
  __shared__ __align__(16) __bf16 sQ[64][64];      // [qrow][hd]
  __shared__ __align__(16) __bf16 sK[64][64];      // [krow][hd]   (= [N][K])
  __shared__ __align__(16) __bf16 sV[64][64];      // [hd][krow]   (= [N][K])
  __shared__ __align__(16) __bf16 sP[4][16][64];   // per-wave probs

  int wave = threadIdx.x >> 5, lane = threadIdx.x & 31;
  int qi = blockIdx.x, head = blockIdx.y, b = blockIdx.z;
  const float scale = 0.125f;  // 1/sqrt(64)
  int hi8 = (lane >> 4) * 8;
  int cl = lane & 15;
  int tr = threadIdx.x >> 1;              // staging row 0..63
  int tc = (threadIdx.x & 1) * 32;        // staging col 0 or 32

  {  // stage Q tile: straight bf16 copy
    const __bf16* src = q + (size_t)(b * SS + qi * 64 + tr) * DD + head * 64 + tc;
#pragma unroll
    for (int i = 0; i < 4; i++)
      *(bf16x8*)&sQ[tr][tc + i * 8] = *(const bf16x8*)(src + i * 8);
  }
  __syncthreads();

  f32x8 ctx[4];
#pragma unroll
  for (int t = 0; t < 4; t++) ctx[t] = (f32x8){0, 0, 0, 0, 0, 0, 0, 0};
  float mrow[8], lrow[8];
#pragma unroll
  for (int r = 0; r < 8; r++) { mrow[r] = -1e30f; lrow[r] = 0.f; }

  for (int j = 0; j < NBK; j++) {
    if (!layout[(head * NBK + qi) * NBK + j]) continue;  // uniform branch
    __syncthreads();  // previous iteration finished with sK/sV
    {  // stage K via async global->LDS (raw 16B copies, no VGPR round-trip)
      const __bf16* srck = k + (size_t)(b * SS + j * 64 + tr) * DD + head * 64 + tc;
#pragma unroll
      for (int i = 0; i < 4; i++)
        async_g2lds_b128(&sK[tr][tc + i * 8], srck + i * 8);
      // stage V transposed (element-level transpose: register path)
      const __bf16* srcv = v + (size_t)(b * SS + j * 64 + tr) * DD + head * 64 + tc;
#pragma unroll
      for (int i = 0; i < 4; i++) {
        bf16x8 fv = *(const bf16x8*)(srcv + i * 8);
#pragma unroll
        for (int t = 0; t < 8; t++) sV[tc + i * 8 + t][tr] = fv[t];
      }
      async_wait0();
    }
    __syncthreads();

    // ---- scores: 16x64 per wave -----------------------------------------
    f32x8 sc[4];
#pragma unroll
    for (int t = 0; t < 4; t++) sc[t] = (f32x8){0, 0, 0, 0, 0, 0, 0, 0};
#pragma unroll
    for (int k0 = 0; k0 < 64; k0 += 32) {
      bf16x16 aq = load_frag(&sQ[0][0], wave * 16, 64, k0, lane);
#pragma unroll
      for (int nt = 0; nt < 4; nt++) {
        bf16x16 bk = load_frag(&sK[0][0], nt * 16, 64, k0, lane);
        sc[nt] = wmma_bf16(aq, bk, sc[nt]);
      }
    }

    // ---- scale + camera bias + mul mask; per-row max --------------------
    int qrow_base = qi * 64 + wave * 16 + hi8;
    int kcol_base = j * 64 + cl;
    float tmax[8];
#pragma unroll
    for (int r = 0; r < 8; r++) tmax[r] = -1e30f;
#pragma unroll
    for (int nt = 0; nt < 4; nt++) {
      int col = kcol_base + nt * 16;
#pragma unroll
      for (int r = 0; r < 8; r++) {
        int row = qrow_base + r;
        float x = sc[nt][r] * scale + bias[((size_t)head * SS + row) * SS + col];
        x *= mask[(size_t)row * SS + col];
        sc[nt][r] = x;
        tmax[r] = fmaxf(tmax[r], x);
      }
    }
#pragma unroll
    for (int m = 8; m >= 1; m >>= 1)
#pragma unroll
      for (int r = 0; r < 8; r++) tmax[r] = fmaxf(tmax[r], __shfl_xor(tmax[r], m, 32));

    // ---- online softmax update ------------------------------------------
    float psum[8];
#pragma unroll
    for (int r = 0; r < 8; r++) {
      float mn = fmaxf(mrow[r], tmax[r]);
      float alpha = __expf(mrow[r] - mn);
      mrow[r] = mn;
      lrow[r] *= alpha;
#pragma unroll
      for (int nt = 0; nt < 4; nt++) ctx[nt][r] *= alpha;
      psum[r] = 0.f;
    }
#pragma unroll
    for (int nt = 0; nt < 4; nt++)
#pragma unroll
      for (int r = 0; r < 8; r++) {
        float p = __expf(sc[nt][r] - mrow[r]);
        psum[r] += p;
        sP[wave][r + hi8][nt * 16 + cl] = (__bf16)p;
      }
#pragma unroll
    for (int m = 8; m >= 1; m >>= 1)
#pragma unroll
      for (int r = 0; r < 8; r++) psum[r] += __shfl_xor(psum[r], m, 32);
#pragma unroll
    for (int r = 0; r < 8; r++) lrow[r] += psum[r];
    __syncthreads();  // publish sP for fragment re-read

    // ---- ctx += P @ V ----------------------------------------------------
#pragma unroll
    for (int k0 = 0; k0 < 64; k0 += 32) {
      bf16x16 ap = load_frag(&sP[wave][0][0], 0, 64, k0, lane);
#pragma unroll
      for (int nt = 0; nt < 4; nt++) {
        bf16x16 bv = load_frag(&sV[0][0], nt * 16, 64, k0, lane);
        ctx[nt] = wmma_bf16(ap, bv, ctx[nt]);
      }
    }
  }

  // ---- h += ctx * (1/l) (residual add; disjoint writes) -------------------
  float rinv[8];
#pragma unroll
  for (int r = 0; r < 8; r++) rinv[r] = 1.0f / lrow[r];
#pragma unroll
  for (int nt = 0; nt < 4; nt++)
#pragma unroll
    for (int r = 0; r < 8; r++) {
      int row = qi * 64 + wave * 16 + r + hi8;
      size_t idx = (size_t)(b * SS + row) * DD + head * 64 + nt * 16 + cl;
      h[idx] += ctx[nt][r] * rinv[r];
    }
}

// ---------------------------------------------------------------------------
// Host-side orchestration
// ---------------------------------------------------------------------------
extern "C" void kernel_launch(void* const* d_in, const int* in_sizes, int n_in,
                              void* d_out, int out_size, void* d_ws,
                              size_t ws_size, hipStream_t stream) {
  (void)in_sizes; (void)n_in; (void)out_size; (void)ws_size;
  const int* x_tokens = (const int*)d_in[0];
  const int* cond_tokens = (const int*)d_in[1];
  const int* layout = (const int*)d_in[2];
  const float* attn_mask = (const float*)d_in[3];
  const float* camera_bias = (const float*)d_in[4];
  const float* x_tok_emb = (const float*)d_in[5];
  const float* cond_tok_emb = (const float*)d_in[6];
  const float* x_pos_emb = (const float*)d_in[7];
  const float* cond_pos_emb = (const float*)d_in[8];
  const float* ln1_g = (const float*)d_in[9];
  const float* ln1_b = (const float*)d_in[10];
  const float* Wq = (const float*)d_in[11];
  const float* bq = (const float*)d_in[12];
  const float* Wk = (const float*)d_in[13];
  const float* bk = (const float*)d_in[14];
  const float* Wv = (const float*)d_in[15];
  const float* bv = (const float*)d_in[16];
  const float* ln2_g = (const float*)d_in[17];
  const float* ln2_b = (const float*)d_in[18];
  const float* W1 = (const float*)d_in[19];
  const float* b1 = (const float*)d_in[20];
  const float* W2 = (const float*)d_in[21];
  const float* b2 = (const float*)d_in[22];
  const float* lnf_g = (const float*)d_in[23];
  const float* lnf_b = (const float*)d_in[24];
  const float* head_W = (const float*)d_in[25];

  // workspace layout (256B aligned slabs)
  char* ws = (char*)d_ws;
  size_t off = 0;
  auto nxt = [&](size_t bytes) -> char* {
    char* p = ws + off;
    off += (bytes + 255) & ~(size_t)255;
    return p;
  };
  float* h = (float*)nxt((size_t)MT * DD * 4);
  float* hn = (float*)nxt((size_t)MT * DD * 4);
  // big slab: MLP mid (f32) overlaid with bf16 q|k|v (disjoint in time)
  char* big = nxt((size_t)MT * DFF * 4);
  float* mid = (float*)big;
  __bf16* qb = (__bf16*)big;
  __bf16* kb = qb + (size_t)MT * DD;
  __bf16* vb = kb + (size_t)MT * DD;
  __bf16* wq_bf = (__bf16*)nxt((size_t)LL * DD * DD * 2);
  __bf16* wk_bf = (__bf16*)nxt((size_t)LL * DD * DD * 2);
  __bf16* wv_bf = (__bf16*)nxt((size_t)LL * DD * DD * 2);
  __bf16* w1_bf = (__bf16*)nxt((size_t)LL * DD * DFF * 2);
  __bf16* w2_bf = (__bf16*)nxt((size_t)LL * DFF * DD * 2);
  __bf16* hw_bf = (__bf16*)nxt((size_t)DD * VV * 2);

  // ---- cast + transpose all weights to bf16 [N][K] ------------------------
  for (int l = 0; l < LL; l++) {
    cast_transpose_kernel<<<(DD * DD + 255) / 256, 256, 0, stream>>>(
        Wq + (size_t)l * DD * DD, wq_bf + (size_t)l * DD * DD, DD, DD);
    cast_transpose_kernel<<<(DD * DD + 255) / 256, 256, 0, stream>>>(
        Wk + (size_t)l * DD * DD, wk_bf + (size_t)l * DD * DD, DD, DD);
    cast_transpose_kernel<<<(DD * DD + 255) / 256, 256, 0, stream>>>(
        Wv + (size_t)l * DD * DD, wv_bf + (size_t)l * DD * DD, DD, DD);
    cast_transpose_kernel<<<(DD * DFF + 255) / 256, 256, 0, stream>>>(
        W1 + (size_t)l * DD * DFF, w1_bf + (size_t)l * DD * DFF, DD, DFF);
    cast_transpose_kernel<<<(DFF * DD + 255) / 256, 256, 0, stream>>>(
        W2 + (size_t)l * DFF * DD, w2_bf + (size_t)l * DFF * DD, DFF, DD);
  }
  cast_transpose_kernel<<<(DD * VV + 255) / 256, 256, 0, stream>>>(head_W, hw_bf, DD, VV);

  // ---- embeddings ---------------------------------------------------------
  embed_kernel<<<MT, 128, 0, stream>>>(x_tokens, cond_tokens, x_tok_emb,
                                       cond_tok_emb, x_pos_emb, cond_pos_emb, h);

  dim3 blk256(256);
  for (int l = 0; l < LL; l++) {
    // h = ln1(h)  (in-place, matches reference)
    ln_kernel<<<MT / 4, 128, 0, stream>>>(h, ln1_g + l * DD, ln1_b + l * DD, h);
    // q,k,v = h @ W + b  -> bf16 [B,S,D] (= [B,S,H,HD] view)
    gemm_kernel<0, 0, true, true><<<dim3(DD / 128, MT / 128), blk256, 0, stream>>>(
        h, wq_bf + (size_t)l * DD * DD, bq + l * DD, qb, MT, DD, DD);
    gemm_kernel<0, 0, true, true><<<dim3(DD / 128, MT / 128), blk256, 0, stream>>>(
        h, wk_bf + (size_t)l * DD * DD, bk + l * DD, kb, MT, DD, DD);
    gemm_kernel<0, 0, true, true><<<dim3(DD / 128, MT / 128), blk256, 0, stream>>>(
        h, wv_bf + (size_t)l * DD * DD, bv + l * DD, vb, MT, DD, DD);
    // h += blocksparse_attention(q,k,v)
    attn_kernel<<<dim3(NBK, HH, BB), 128, 0, stream>>>(qb, kb, vb, layout,
                                                       camera_bias, attn_mask, h);
    // hn = ln2(h); h += gelu(hn@W1+b1)@W2 + b2
    ln_kernel<<<MT / 4, 128, 0, stream>>>(h, ln2_g + l * DD, ln2_b + l * DD, hn);
    gemm_kernel<1, 0, true, false><<<dim3(DFF / 128, MT / 128), blk256, 0, stream>>>(
        hn, w1_bf + (size_t)l * DD * DFF, b1 + l * DFF, mid, MT, DFF, DD);
    gemm_kernel<0, 1, true, false><<<dim3(DD / 128, MT / 128), blk256, 0, stream>>>(
        mid, w2_bf + (size_t)l * DFF * DD, b2 + l * DD, h, MT, DD, DFF);
  }
  // final LN + vocab head
  ln_kernel<<<MT / 4, 128, 0, stream>>>(h, lnf_g, lnf_b, hn);
  gemm_kernel<0, 0, false, false><<<dim3(VV / 128, MT / 128), blk256, 0, stream>>>(
      hn, hw_bf, nullptr, (float*)d_out, MT, VV, DD);
}
